// Conv2D_1632087572949
// MI455X (gfx1250) — matrix operand order repaired and checked by reading, not compiled, not run on hardware
//
#include <hip/hip_runtime.h>

// Conv2D 15x15 VALID on 4096x4096 fp32, via Toeplitz-GEMM on CDNA5 WMMA.
// f16 hi/lo split (3 WMMAs: xh*wh + xh*wl + xl*wh) drops only the xl*wl term
// (~2^-22 relative) -> fp32-grade accuracy on the confirmed
// v_wmma_f32_16x16x32_f16 path.

typedef __attribute__((ext_vector_type(16))) _Float16 v16h;
typedef __attribute__((ext_vector_type(8)))  _Float16 v8h;
typedef __attribute__((ext_vector_type(8)))  float    v8f;

#define IH 4096
#define IW 4096
#define KH 15
#define KW 15
#define OH (IH - KH + 1)   // 4082
#define OW (IW - KW + 1)   // 4082

#define WG_OY 32           // outputs per workgroup in y (2 wave-tiles)
#define WG_OX 64           // outputs per workgroup in x (4 wave-tiles)
#define IN_R  46           // staged input rows:  32 + 14
#define IN_C  80           // staged input cols:  48 + 32 (K=32 window of last tile)
#define IN_S  88           // LDS row stride (f16), multiple of 8 -> 16B-aligned rows
#define WIN   (16 + KW - 1) // 30: valid Toeplitz window width

__global__ __launch_bounds__(256)
void conv15_wmma_f16x3(const float* __restrict__ x,
                       const float* __restrict__ wgt,
                       const float* __restrict__ bias,
                       float* __restrict__ out)
{
    __shared__ __align__(32) _Float16 sAh[IN_R * IN_S];
    __shared__ __align__(32) _Float16 sAl[IN_R * IN_S];
    __shared__ __align__(32) _Float16 sBh[KH * 512];   // per i: 32 lanes x 16 f16
    __shared__ __align__(32) _Float16 sBl[KH * 512];

    const int tid = threadIdx.x;
    const int gx0 = blockIdx.x * WG_OX;
    const int gy0 = blockIdx.y * WG_OY;

    // ---- Stage input halo (clamped, always in-bounds & finite) as f16 hi/lo ----
    for (int idx = tid; idx < IN_R * IN_C; idx += 256) {
        int r = idx / IN_C;
        int c = idx - r * IN_C;
        int gy = gy0 + r; if (gy > IH - 1) gy = IH - 1;
        int gx = gx0 + c; if (gx > IW - 1) gx = IW - 1;
        float v = x[(size_t)gy * IW + gx];
        _Float16 hi = (_Float16)v;
        sAh[r * IN_S + c] = hi;
        sAl[r * IN_S + c] = (_Float16)(v - (float)hi);
    }

    // ---- Precompute Toeplitz B fragments in WMMA per-lane layout ----
    // B_i[k][n] = w(i, k-n) if 0 <= k-n < 15 and k < 30, else 0.
    // Lane L = p>>4, elem e = p&15; row k = e + 16*(L>>4); col n = L&15.
    for (int idx = tid; idx < KH * 512; idx += 256) {
        int i    = idx >> 9;
        int p    = idx & 511;
        int lane = p >> 4;
        int e    = p & 15;
        int k    = e + ((lane >> 4) << 4);
        int n    = lane & 15;
        int j    = k - n;
        float wv = 0.0f;
        if (k < WIN && j >= 0 && j < KW) wv = wgt[i * KW + j];
        _Float16 hi = (_Float16)wv;
        sBh[idx] = hi;
        sBl[idx] = (_Float16)(wv - (float)hi);
    }
    __syncthreads();

    // ---- Per-wave 16x16 output tile ----
    const int lane  = tid & 31;
    const int wave  = tid >> 5;
    const int ty    = wave >> 2;        // 0..1
    const int tx    = wave & 3;         // 0..3
    const int hh    = lane >> 4;        // wave half
    const int m     = lane & 15;        // A row within tile
    const int tyOff = ty << 4;
    const int txOff = tx << 4;

    v8f acc = {0.f, 0.f, 0.f, 0.f, 0.f, 0.f, 0.f, 0.f};

    #pragma unroll
    for (int i = 0; i < KH; ++i) {
        // A 16-bit layout: lane half hh, elems 0..15 -> K = 16*hh..16*hh+15
        // (contiguous, matching B; two 16B runs per lane).
        const _Float16* rowH = &sAh[(tyOff + m + i) * IN_S + txOff];
        const _Float16* rowL = &sAl[(tyOff + m + i) * IN_S + txOff];
        v8h ah0 = *(const v8h*)(rowH + 16 * hh);
        v8h ah1 = *(const v8h*)(rowH + 16 * hh + 8);
        v8h al0 = *(const v8h*)(rowL + 16 * hh);
        v8h al1 = *(const v8h*)(rowL + 16 * hh + 8);
        v16h ah = __builtin_shufflevector(ah0, ah1, 0,1,2,3,4,5,6,7,8,9,10,11,12,13,14,15);
        v16h al = __builtin_shufflevector(al0, al1, 0,1,2,3,4,5,6,7,8,9,10,11,12,13,14,15);
        v16h bh = *(const v16h*)&sBh[i * 512 + lane * 16];
        v16h bl = *(const v16h*)&sBl[i * 512 + lane * 16];

        // f16x3 split: x*w ~= xh*wh + xh*wl + xl*wh  (xl*wl ~ 2^-22, dropped)
        acc = __builtin_amdgcn_wmma_f32_16x16x32_f16(false, ah, false, bh, (short)0, acc, false, false);
        acc = __builtin_amdgcn_wmma_f32_16x16x32_f16(false, ah, false, bl, (short)0, acc, false, false);
        acc = __builtin_amdgcn_wmma_f32_16x16x32_f16(false, al, false, bh, (short)0, acc, false, false);
    }

    // ---- Store: C/D layout VGPR v -> row v + 8*hh, col = lane&15 ----
    const float bv = bias[0];
    const int ox = gx0 + txOff + m;
    if (ox < OW) {
        #pragma unroll
        for (int v = 0; v < 8; ++v) {
            int oy = gy0 + tyOff + 8 * hh + v;
            if (oy < OH) out[(size_t)oy * OW + ox] = acc[v] + bv;
        }
    }
}

extern "C" void kernel_launch(void* const* d_in, const int* in_sizes, int n_in,
                              void* d_out, int out_size, void* d_ws, size_t ws_size,
                              hipStream_t stream) {
    const float* x    = (const float*)d_in[0];   // 4096*4096 fp32
    const float* wgt  = (const float*)d_in[1];   // 15*15 fp32
    const float* bias = (const float*)d_in[2];   // 1 fp32
    float* out = (float*)d_out;                  // 4082*4082 fp32

    dim3 block(256, 1, 1);
    dim3 grid((OW + WG_OX - 1) / WG_OX,          // 64
              (OH + WG_OY - 1) / WG_OY,          // 128
              1);
    conv15_wmma_f16x3<<<grid, block, 0, stream>>>(x, wgt, bias, out);
}
